// SceneHGT_13116830122417
// MI455X (gfx1250) — matrix operand order repaired and checked by reading, not compiled, not run on hardware
//
#include <hip/hip_runtime.h>
#include <cmath>

// ---------------- problem constants (fixed by reference) ----------------
static constexpr int N_  = 20000;   // nodes
static constexpr int C_  = 256;     // channels
static constexpr int H_  = 8;       // heads
static constexpr int DH_ = 32;      // head dim
static constexpr int E_  = 160000;  // edges per relation
static constexpr int R_  = 2;       // relations
static constexpr int C3_ = 768;     // 3*C

typedef __bf16 bf16;
typedef __bf16 bf16x8  __attribute__((ext_vector_type(8)));
typedef __bf16 bf16x16 __attribute__((ext_vector_type(16)));
typedef float  f32x8   __attribute__((ext_vector_type(8)));

union ABfrag { bf16x16 v; bf16x8 h[2]; };

__device__ __forceinline__ f32x8 wmma_bf16(bf16x16 a, bf16x16 b, f32x8 c) {
  // D = A(16x32 bf16) x B(32x16 bf16) + C(16x16 f32)
  return __builtin_amdgcn_wmma_f32_16x16x32_bf16(false, a, false, b, (short)0, c,
                                                 false, false);
}

// monotonic float<->uint encoding for atomicMax on signed floats
__device__ __forceinline__ unsigned encf(float x) {
  unsigned u = __float_as_uint(x);
  return (u & 0x80000000u) ? ~u : (u | 0x80000000u);
}
__device__ __forceinline__ float decf(unsigned u) {
  return (u & 0x80000000u) ? __uint_as_float(u & 0x7FFFFFFFu)
                           : __uint_as_float(~u);
}
static constexpr unsigned ENC_NEG_INF = 0x007FFFFFu; // encf(-inf)

// ---------------- small conversion kernels ----------------
__global__ void k_x2bf(const float* __restrict__ x, bf16* __restrict__ hbf) {
  int idx = blockIdx.x * 256 + threadIdx.x;
  if (idx < N_ * C_) hbf[idx] = (bf16)x[idx];
}

// kqv_w[k][n] (256x768) -> WT[n][k]  (column-of-B contiguous in k)
__global__ void k_wcvt_kqv(const float* __restrict__ W, bf16* __restrict__ WT) {
  int idx = blockIdx.x * 256 + threadIdx.x;   // < 768*256
  int n = idx >> 8, k = idx & 255;
  WT[idx] = (bf16)W[k * C3_ + n];
}
// out_w[k][n] (256x256) -> WT[n][k]
__global__ void k_wcvt_out(const float* __restrict__ W, bf16* __restrict__ WT) {
  int idx = blockIdx.x * 256 + threadIdx.x;   // < 256*256
  int n = idx >> 8, k = idx & 255;
  WT[idx] = (bf16)W[k * C_ + n];
}
// krel[r][h][d][f] copies as-is (B-ready for qrel GEMM: lane d loads krel[d][:])
// vrel[r][h][d][f] -> vT[r][h][f][d]   (B-ready for vtr GEMM)
__global__ void k_wcvt_rel(const float* __restrict__ kr, const float* __restrict__ vr,
                           bf16* __restrict__ kbf, bf16* __restrict__ vTbf) {
  int idx = blockIdx.x * 256 + threadIdx.x;   // < R*H*32*32 = 16384
  kbf[idx] = (bf16)kr[idx];
  int rh = idx >> 10;
  int d  = (idx >> 5) & 31;
  int f  = idx & 31;
  vTbf[(rh << 10) + f * 32 + d] = (bf16)vr[idx];
}

__global__ void k_init(float* __restrict__ agg, unsigned* __restrict__ mx,
                       float* __restrict__ sm) {
  int idx = blockIdx.x * 256 + threadIdx.x;
  if (idx < N_ * C_) agg[idx] = 0.f;
  if (idx < N_ * H_) { mx[idx] = ENC_NEG_INF; sm[idx] = 0.f; }
}

// ---------------- GEMM 1: kqv = h @ kqv_w + b  (bf16 WMMA, f32 acc) ----------------
// A: hbf [N x 256], B: WT [768 x 256] (col-ready), out: kqvbf [N x 768] bf16
__global__ void k_gemm_kqv(const bf16* __restrict__ A, const bf16* __restrict__ Bc,
                           const float* __restrict__ bias, bf16* __restrict__ O) {
  int lane = threadIdx.x & 31, w = threadIdx.x >> 5;   // 8 waves
  int wm = w & 1, wn = w >> 1;                         // 2 x 4 wave grid
  int row0 = blockIdx.x * 32 + wm * 16;
  int col0 = blockIdx.y * 64 + wn * 16;
  int m = lane & 15, half = lane >> 4;
  const bf16* arow = A + (size_t)(row0 + m) * C_;
  const bf16* bcol = Bc + (size_t)(col0 + m) * C_ + half * 16;
  f32x8 acc = {0.f,0.f,0.f,0.f,0.f,0.f,0.f,0.f};
  for (int kk = 0; kk < C_; kk += 32) {
    ABfrag af, bfr;
    af.h[0]  = *(const bf16x8*)(arow + kk + half * 16);
    af.h[1]  = *(const bf16x8*)(arow + kk + half * 16 + 8);
    bfr.h[0] = *(const bf16x8*)(bcol + kk);
    bfr.h[1] = *(const bf16x8*)(bcol + kk + 8);
    acc = wmma_bf16(af.v, bfr.v, acc);
  }
  int gc = col0 + m;
  float b = bias[gc];
#pragma unroll
  for (int j = 0; j < 8; ++j) {
    int rr = row0 + j + half * 8;
    O[(size_t)rr * C3_ + gc] = (bf16)(acc[j] + b);
  }
}

// ---------------- batched per-(r,h) 32x32 relation transforms ----------------
// z==0: qrel[r,n,h,:] = q[n,h,:] @ krel[r,h]^T   (B = krel as stored)
// z==1: vtr [r,n,h,:] = v[n,h,:] @ vrel[r,h]     (B = vrelT)
__global__ void k_rel(const bf16* __restrict__ kqv, const bf16* __restrict__ krbf,
                      const bf16* __restrict__ vrbf, bf16* __restrict__ qrel,
                      bf16* __restrict__ vtr) {
  int lane = threadIdx.x & 31, ntile = threadIdx.x >> 5;  // 2 waves
  int rh = blockIdx.y;              // r*H + h
  int which = blockIdx.z;           // 0 -> qrel, 1 -> vtr
  int r = rh >> 3, h = rh & 7;
  int row0 = blockIdx.x * 16;
  int m = lane & 15, half = lane >> 4;
  const bf16* Asrc = kqv + (size_t)(row0 + m) * C3_ + (which ? 512 : 256) + h * DH_;
  const bf16* W = (which ? vrbf : krbf) + rh * DH_ * DH_;
  int dcol = ntile * 16 + m;
  ABfrag af, bfr;
  af.h[0] = *(const bf16x8*)(Asrc + half * 16);
  af.h[1] = *(const bf16x8*)(Asrc + half * 16 + 8);
  const bf16* bcol = W + dcol * DH_ + half * 16;
  bfr.h[0] = *(const bf16x8*)(bcol);
  bfr.h[1] = *(const bf16x8*)(bcol + 8);
  f32x8 acc = {0.f,0.f,0.f,0.f,0.f,0.f,0.f,0.f};
  acc = wmma_bf16(af.v, bfr.v, acc);
  bf16* O = (which ? vtr : qrel) + (size_t)(r * N_) * C_;
#pragma unroll
  for (int j = 0; j < 8; ++j) {
    int n = row0 + j + half * 8;
    O[(size_t)n * C_ + h * DH_ + dcol] = (bf16)acc[j];
  }
}

// ---------------- edge pass 1: logits + segment max ----------------
__global__ void k_logits(const int* __restrict__ ei0, const int* __restrict__ ei1,
                         const bf16* __restrict__ kqv, const bf16* __restrict__ qrel,
                         const float* __restrict__ prel, int layer,
                         float* __restrict__ ew, unsigned* __restrict__ mx) {
  int idx = blockIdx.x * 256 + threadIdx.x;     // < 2E*H
  int eg = idx >> 3, h = idx & 7;
  int r = (eg >= E_) ? 1 : 0;
  int e = eg - r * E_;
  const int* ei = r ? ei1 : ei0;
  int src = ei[e], dst = ei[E_ + e];
  const bf16* kp = kqv + (size_t)src * C3_ + h * DH_;            // k block (cols 0..255)
  const bf16* qp = qrel + (size_t)(r * N_ + dst) * C_ + h * DH_;
  float acc = 0.f;
#pragma unroll
  for (int d = 0; d < DH_; ++d) acc += (float)kp[d] * (float)qp[d];
  float l = acc * prel[layer * R_ * H_ + r * H_ + h] * 0.17677669529663689f; // 1/sqrt(32)
  ew[idx] = l;
  atomicMax(mx + dst * H_ + h, encf(l));
}

// ---------------- edge pass 2: exp + segment sum (+ prefetch vtr for pass 3) ----
__global__ void k_expsum(const int* __restrict__ ei0, const int* __restrict__ ei1,
                         const unsigned* __restrict__ mx, float* __restrict__ ew,
                         float* __restrict__ sm, const bf16* __restrict__ vtr) {
  int idx = blockIdx.x * 256 + threadIdx.x;
  int eg = idx >> 3, h = idx & 7;
  int r = (eg >= E_) ? 1 : 0;
  int e = eg - r * E_;
  const int* ei = r ? ei1 : ei0;
  int src = ei[e], dst = ei[E_ + e];
  float m = decf(mx[dst * H_ + h]);
  float ev = expf(ew[idx] - m);
  ew[idx] = ev;
  atomicAdd(sm + dst * H_ + h, ev);
  // warm L2 with the message line the aggregate pass will gather
  __builtin_prefetch(vtr + (size_t)(r * N_ + src) * C_ + h * DH_, 0, 1);
}

// ---------------- edge pass 3: normalize + scatter-aggregate ----------------
__global__ void k_agg(const int* __restrict__ ei0, const int* __restrict__ ei1,
                      const float* __restrict__ ew, const float* __restrict__ sm,
                      const bf16* __restrict__ vtr, float* __restrict__ agg) {
  int idx = blockIdx.x * 256 + threadIdx.x;
  int eg = idx >> 3, h = idx & 7;
  int r = (eg >= E_) ? 1 : 0;
  int e = eg - r * E_;
  const int* ei = r ? ei1 : ei0;
  int src = ei[e], dst = ei[E_ + e];
  float alpha = ew[idx] / fmaxf(sm[dst * H_ + h], 1e-16f);
  const bf16* vp = vtr + (size_t)(r * N_ + src) * C_ + h * DH_;
  float* ap = agg + (size_t)dst * C_ + h * DH_;
#pragma unroll
  for (int d = 0; d < DH_; ++d) atomicAdd(ap + d, alpha * (float)vp[d]);
}

// ---------------- exact GELU -> bf16 A-operand ----------------
__global__ void k_gelu(const float* __restrict__ agg, bf16* __restrict__ gbf) {
  int idx = blockIdx.x * 256 + threadIdx.x;
  if (idx >= N_ * C_) return;
  float a = agg[idx];
  gbf[idx] = (bf16)(0.5f * a * (1.f + erff(a * 0.70710678118654752f)));
}

// ---------------- GEMM 2: out proj + sigmoid-gated skip ----------------
__global__ void k_gemm_out(const bf16* __restrict__ A, const bf16* __restrict__ Bc,
                           const float* __restrict__ bias, const float* __restrict__ skip,
                           int layer, const float* __restrict__ hold,
                           float* __restrict__ hout) {
  int lane = threadIdx.x & 31, w = threadIdx.x >> 5;
  int wm = w & 1, wn = w >> 1;
  int row0 = blockIdx.x * 32 + wm * 16;
  int col0 = blockIdx.y * 64 + wn * 16;
  int m = lane & 15, half = lane >> 4;
  const bf16* arow = A + (size_t)(row0 + m) * C_;
  const bf16* bcol = Bc + (size_t)(col0 + m) * C_ + half * 16;
  f32x8 acc = {0.f,0.f,0.f,0.f,0.f,0.f,0.f,0.f};
  for (int kk = 0; kk < C_; kk += 32) {
    ABfrag af, bfr;
    af.h[0]  = *(const bf16x8*)(arow + kk + half * 16);
    af.h[1]  = *(const bf16x8*)(arow + kk + half * 16 + 8);
    bfr.h[0] = *(const bf16x8*)(bcol + kk);
    bfr.h[1] = *(const bf16x8*)(bcol + kk + 8);
    acc = wmma_bf16(af.v, bfr.v, acc);
  }
  int gc = col0 + m;
  float b = bias[gc];
  float s = 1.f / (1.f + expf(-skip[layer]));
#pragma unroll
  for (int j = 0; j < 8; ++j) {
    int rr = row0 + j + half * 8;
    size_t o = (size_t)rr * C_ + gc;
    hout[o] = s * (acc[j] + b) + (1.f - s) * hold[o];
  }
}

// ---------------- ReLU + LayerNorm (+ bf16 A-operand for next layer) ----------
__global__ void k_ln(float* __restrict__ h, bf16* __restrict__ hbf,
                     const float* __restrict__ g, const float* __restrict__ b) {
  int warp = threadIdx.x >> 5, lane = threadIdx.x & 31;
  int row = blockIdx.x * 8 + warp;
  const float* hr = h + (size_t)row * C_;
  float v[8];
  float s = 0.f;
#pragma unroll
  for (int j = 0; j < 8; ++j) {
    float x = fmaxf(hr[lane * 8 + j], 0.f);   // ReLU before LN (matches reference)
    v[j] = x; s += x;
  }
  for (int off = 16; off > 0; off >>= 1) s += __shfl_xor(s, off, 32);
  float mu = s * (1.f / 256.f);
  float var = 0.f;
#pragma unroll
  for (int j = 0; j < 8; ++j) { float d = v[j] - mu; var += d * d; }
  for (int off = 16; off > 0; off >>= 1) var += __shfl_xor(var, off, 32);
  var *= (1.f / 256.f);
  float inv = rsqrtf(var + 1e-5f);
  float* hw = h + (size_t)row * C_;
#pragma unroll
  for (int j = 0; j < 8; ++j) {
    int c = lane * 8 + j;
    float y = (v[j] - mu) * inv * g[c] + b[c];
    hw[c] = y;
    hbf[(size_t)row * C_ + c] = (bf16)y;
  }
}

// ---------------- host orchestration ----------------
extern "C" void kernel_launch(void* const* d_in, const int* in_sizes, int n_in,
                              void* d_out, int out_size, void* d_ws, size_t ws_size,
                              hipStream_t stream) {
  (void)in_sizes; (void)n_in; (void)out_size; (void)ws_size;
  const float* x     = (const float*)d_in[0];
  const int*   ei0   = (const int*)d_in[1];
  const int*   ei1   = (const int*)d_in[2];
  const float* kqv_w = (const float*)d_in[3];
  const float* kqv_b = (const float*)d_in[4];
  const float* out_w = (const float*)d_in[5];
  const float* out_b = (const float*)d_in[6];
  const float* skip  = (const float*)d_in[7];
  const float* krel  = (const float*)d_in[8];
  const float* vrel  = (const float*)d_in[9];
  const float* prel  = (const float*)d_in[10];
  const float* ln_g  = (const float*)d_in[11];
  const float* ln_b  = (const float*)d_in[12];
  float* out = (float*)d_out;

  // workspace carve-out (256B aligned)
  char* p = (char*)d_ws;
  auto carve = [&](size_t bytes) -> char* {
    char* q = p; p += (bytes + 255) & ~size_t(255); return q;
  };
  bf16*     hbf   = (bf16*)    carve((size_t)N_ * C_ * 2);        // current h (bf16 A)
  bf16*     kqvbf = (bf16*)    carve((size_t)N_ * C3_ * 2);       // fused k|q|v
  bf16*     qrel  = (bf16*)    carve((size_t)R_ * N_ * C_ * 2);   // krel^T @ q per node
  bf16*     vtr   = (bf16*)    carve((size_t)R_ * N_ * C_ * 2);   // vrel-transformed v
  float*    ew    = (float*)   carve((size_t)2 * E_ * H_ * 4);    // logits -> exp
  unsigned* mx    = (unsigned*)carve((size_t)N_ * H_ * 4);        // encoded seg max
  float*    sm    = (float*)   carve((size_t)N_ * H_ * 4);        // seg sum
  float*    agg   = (float*)   carve((size_t)N_ * C_ * 4);        // aggregated msgs
  bf16*     gbf   = (bf16*)    carve((size_t)N_ * C_ * 2);        // gelu(agg) bf16
  float*    hf    = (float*)   carve((size_t)N_ * C_ * 4);        // f32 hidden state
  bf16*     wKT   = (bf16*)    carve((size_t)C3_ * C_ * 2);       // kqv_w^T bf16
  bf16*     wOT   = (bf16*)    carve((size_t)C_ * C_ * 2);        // out_w^T bf16
  bf16*     krbf  = (bf16*)    carve((size_t)R_ * H_ * DH_ * DH_ * 2);
  bf16*     vrbf  = (bf16*)    carve((size_t)R_ * H_ * DH_ * DH_ * 2);

  k_x2bf<<<(N_ * C_ + 255) / 256, 256, 0, stream>>>(x, hbf);

  const int relSz = R_ * H_ * DH_ * DH_;
  const int edgeBlocks = (2 * E_ * H_) / 256;

  for (int i = 0; i < 4; ++i) {
    k_wcvt_kqv<<<(C3_ * C_) / 256, 256, 0, stream>>>(kqv_w + (size_t)i * C_ * C3_, wKT);
    k_wcvt_out<<<(C_ * C_) / 256, 256, 0, stream>>>(out_w + (size_t)i * C_ * C_, wOT);
    k_wcvt_rel<<<relSz / 256, 256, 0, stream>>>(krel + (size_t)i * relSz,
                                                vrel + (size_t)i * relSz, krbf, vrbf);
    k_init<<<(N_ * C_ + 255) / 256, 256, 0, stream>>>(agg, mx, sm);

    dim3 g1(N_ / 32, C3_ / 64);
    k_gemm_kqv<<<g1, 256, 0, stream>>>(hbf, wKT, kqv_b + i * C3_, kqvbf);

    dim3 g2(N_ / 16, R_ * H_, 2);
    k_rel<<<g2, 64, 0, stream>>>(kqvbf, krbf, vrbf, qrel, vtr);

    k_logits<<<edgeBlocks, 256, 0, stream>>>(ei0, ei1, kqvbf, qrel, prel, i, ew, mx);
    k_expsum<<<edgeBlocks, 256, 0, stream>>>(ei0, ei1, mx, ew, sm, vtr);
    k_agg<<<edgeBlocks, 256, 0, stream>>>(ei0, ei1, ew, sm, vtr, agg);

    k_gelu<<<(N_ * C_) / 256, 256, 0, stream>>>(agg, gbf);

    const float* hold = (i == 0) ? x : hf;
    float* hdst = (i == 3) ? out : hf;
    dim3 g3(N_ / 32, C_ / 64);
    k_gemm_out<<<g3, 256, 0, stream>>>(gbf, wOT, out_b + i * C_, skip, i, hold, hdst);

    if (i < 3)
      k_ln<<<N_ / 8, 256, 0, stream>>>(hf, hbf, ln_g + i * C_, ln_b + i * C_);
  }
}
